// _RaggedBlock_15358803051124
// MI455X (gfx1250) — matrix operand order repaired and checked by reading, not compiled, not run on hardware
//
#include <hip/hip_runtime.h>

// ---------------------------------------------------------------------------
// Types for CDNA5 WMMA (wave32, 16x16x32 bf16 -> f32)
// ---------------------------------------------------------------------------
typedef unsigned short u16;
typedef __bf16 bf16_t;
typedef bf16_t bf16x16 __attribute__((ext_vector_type(16)));
typedef float  floatx8 __attribute__((ext_vector_type(8)));
typedef unsigned int uint32x4 __attribute__((ext_vector_type(4)));
typedef int int32x8 __attribute__((ext_vector_type(8)));
typedef int int32x4 __attribute__((ext_vector_type(4)));

#if defined(__has_builtin)
#  if __has_builtin(__builtin_amdgcn_tensor_load_to_lds) && \
      __has_builtin(__builtin_amdgcn_s_wait_tensorcnt)
#    define USE_TDM 1
#  endif
#endif
#ifndef USE_TDM
#  define USE_TDM 0
#endif

union Frag {                 // 8 VGPRs = 16 bf16 elements (A or B fragment)
    unsigned int u32[8];
    bf16x16 v;
};

__device__ __forceinline__ u16 f2bf(float f) {
    unsigned int u = __float_as_uint(f);
    unsigned int r = u + 0x7FFFu + ((u >> 16) & 1u);   // round-to-nearest-even
    return (u16)(r >> 16);
}

#define WMMA_BF16(A, B, C) \
    __builtin_amdgcn_wmma_f32_16x16x32_bf16(false, (A), false, (B), (short)0, (C), false, false)

#if USE_TDM
// ---------------------------------------------------------------------------
// TDM: 2D tile load global->LDS. tile = tile_d1 rows x tile_d0 bf16 elements.
// LDS padding: after every 16 dwords (one 32-elem row) insert 4 dwords
// -> LDS row stride 40 bf16 elements (matches sA layout below).
// Descriptor bitfields per CDNA5 ISA 8.3/8.4 (D# group 0 / group 1).
// This toolchain's builtin takes 6 args: (g0 v4u32, g1 v8i32, g2 v4i32,
// g3 v4i32, v8i32, i32 cpol) -- groups 2/3 unused for a 2D tile.
// ---------------------------------------------------------------------------
__device__ __forceinline__ void tdm_load_2d_bf16(
    const u16* gaddr, unsigned lds_off,
    int tensor_d0, int tensor_d1,            // remaining tensor extent (elems)
    int tile_d0, int tile_d1,                // tile size (elems); tile_d0 = 32
    long long stride0)                       // row stride (elems)
{
    unsigned long long ga = (unsigned long long)(size_t)gaddr;
    uint32x4 g0;
    g0[0] = 0x1u;                                        // count=1, user mode
    g0[1] = lds_off;                                     // lds_addr (bytes)
    g0[2] = (unsigned)(ga & 0xffffffffu);                // global_addr[31:0]
    g0[3] = (unsigned)((ga >> 32) & 0x1ffffffu)          // global_addr[56:32]
          | (2u << 30);                                  // type = 2 (image)
    int32x8 g1;
    g1[0] = (1 << 16)                                    // data_size = 2 bytes
          | (1 << 20)                                    // pad_enable
          | (3 << 22)                                    // pad_interval: 16 dwords
          | (3 << 25);                                   // pad_amount: 4 dwords
    g1[1] = (tensor_d0 & 0xffff) << 16;                  // tensor_dim0[15:0]
    g1[2] = ((tensor_d0 >> 16) & 0xffff)                 // tensor_dim0[31:16]
          | ((tensor_d1 & 0xffff) << 16);                // tensor_dim1[15:0]
    g1[3] = ((tensor_d1 >> 16) & 0xffff)                 // tensor_dim1[31:16]
          | ((tile_d0 & 0xffff) << 16);                  // tile_dim0
    g1[4] = (tile_d1 & 0xffff);                          // tile_dim1 (tile_dim2=0)
    g1[5] = (int)(stride0 & 0xffffffffll);               // tensor_dim0_stride[31:0]
    g1[6] = (int)((stride0 >> 32) & 0xffffll);           // stride[47:32] (dim1_stride=0)
    g1[7] = 0;
    int32x4 z4 = {0, 0, 0, 0};
    int32x8 z8 = {0, 0, 0, 0, 0, 0, 0, 0};
    __builtin_amdgcn_tensor_load_to_lds(g0, g1, z4, z4, z8, 0);
}
#endif

// ---------------------------------------------------------------------------
// fp32 -> bf16 weight conversion
// ---------------------------------------------------------------------------
__global__ void __launch_bounds__(256) cvt_bf16_kernel(const float* __restrict__ in,
                                                       u16* __restrict__ out, int n) {
    int i = blockIdx.x * 256 + threadIdx.x;
    if (i < n) out[i] = f2bf(in[i]);
}

// ---------------------------------------------------------------------------
// LayerNorm over D=768, one row per 256-thread block, bf16 output
// ---------------------------------------------------------------------------
__global__ void __launch_bounds__(256) ln_kernel(const float* __restrict__ x,
                                                 const float* __restrict__ g,
                                                 const float* __restrict__ be,
                                                 u16* __restrict__ out) {
    const int D = 768;
    int row = blockIdx.x, tid = threadIdx.x;
    const float* xr = x + (size_t)row * D;
    float v0 = xr[tid], v1 = xr[tid + 256], v2 = xr[tid + 512];

    __shared__ float red[256];
    red[tid] = v0 + v1 + v2;
    __syncthreads();
    for (int off = 128; off > 0; off >>= 1) {
        if (tid < off) red[tid] += red[tid + off];
        __syncthreads();
    }
    float mu = red[0] * (1.0f / 768.0f);
    __syncthreads();
    float d0 = v0 - mu, d1 = v1 - mu, d2 = v2 - mu;
    red[tid] = d0 * d0 + d1 * d1 + d2 * d2;
    __syncthreads();
    for (int off = 128; off > 0; off >>= 1) {
        if (tid < off) red[tid] += red[tid + off];
        __syncthreads();
    }
    float rstd = rsqrtf(red[0] * (1.0f / 768.0f) + 1e-6f);

    u16* orow = out + (size_t)row * D;
    orow[tid]       = f2bf(d0 * rstd * g[tid]       + be[tid]);
    orow[tid + 256] = f2bf(d1 * rstd * g[tid + 256] + be[tid + 256]);
    orow[tid + 512] = f2bf(d2 * rstd * g[tid + 512] + be[tid + 512]);
}

// ---------------------------------------------------------------------------
// Generic bf16 WMMA GEMM: C = A(MxK) * B(KxN) + bias [+ res] [gelu]
// 8 waves / block, block tile 128x128, K-step 32.
// A tile staged by the Tensor Data Mover (wave 0 issues, s_wait_tensorcnt),
// B tile staged transposed by all threads.
// ---------------------------------------------------------------------------
enum { EPI_BF16 = 0, EPI_F32_RES = 1, EPI_GELU_BF16 = 2 };

template <int EPI>
__global__ void __launch_bounds__(256) gemm_bf16_kernel(
    const u16* __restrict__ A, const u16* __restrict__ Bm,
    const float* __restrict__ bias, const float* __restrict__ res,
    u16* __restrict__ outB, float* __restrict__ outF,
    int M, int N, int K) {
    const int lda = K, ldb = N;
    __shared__ u16 sA[128 * 40];    // [m][k], stride 40 (80B, TDM-padded rows)
    __shared__ u16 sBt[128 * 36];   // [n][k], stride 36 -> 4B aligned dword reads

    int tid = threadIdx.x;
    int bm = blockIdx.y, bn = blockIdx.x;
    int wid = tid >> 5, lane = tid & 31;
    int mo = (wid >> 2) * 64, no = (wid & 3) * 32;     // wave tile: 64x32
    int r = lane & 15, hi = lane >> 4;
    int kb8 = hi * 8, kb16 = hi * 16;

#if USE_TDM
    unsigned sA_lds = (unsigned)(size_t)(void*)&sA[0];
    int rowsLeft = M - bm * 128;
#endif

    floatx8 acc[4][2];
#pragma unroll
    for (int mi = 0; mi < 4; mi++)
#pragma unroll
        for (int ni = 0; ni < 2; ni++)
#pragma unroll
            for (int v = 0; v < 8; v++) acc[mi][ni][v] = 0.0f;

    for (int k0 = 0; k0 < K; k0 += 32) {
        // ---- A tile: 128 rows x 32 k
#if USE_TDM
        if (wid == 0) {
            tdm_load_2d_bf16(A + (size_t)(bm * 128) * lda + k0, sA_lds,
                             K - k0, rowsLeft, 32, 128, (long long)lda);
        }
#else
#pragma unroll
        for (int it = 0; it < 2; ++it) {
            int c = tid + it * 256;
            int row = c >> 2, kc = (c & 3) * 8;
            int gm = bm * 128 + row; if (gm >= M) gm = M - 1;
            *(uint4*)&sA[row * 40 + kc] =
                *(const uint4*)(A + (size_t)gm * lda + k0 + kc);
        }
#endif
        // ---- B tile: 32 k x 128 n, stored transposed
#pragma unroll
        for (int it = 0; it < 2; ++it) {
            int c = tid + it * 256;                    // 512 chunks of 8 bf16
            int kr = c >> 4, nc = (c & 15) * 8;
            uint4 tb = *(const uint4*)(Bm + (size_t)(k0 + kr) * ldb + bn * 128 + nc);
            unsigned int w[4] = {tb.x, tb.y, tb.z, tb.w};
#pragma unroll
            for (int i = 0; i < 4; i++) {
                sBt[(nc + 2 * i) * 36 + kr]     = (u16)(w[i] & 0xffffu);
                sBt[(nc + 2 * i + 1) * 36 + kr] = (u16)(w[i] >> 16);
            }
        }
        if (k0 + 32 < K)
            __builtin_prefetch(Bm + (size_t)(k0 + 32) * ldb + bn * 128, 0, 1);
#if USE_TDM
        if (wid == 0) __builtin_amdgcn_s_wait_tensorcnt(0);
#endif
        __syncthreads();

        Frag a[4], b[2];
#pragma unroll
        for (int mi = 0; mi < 4; mi++)
#pragma unroll
            for (int v = 0; v < 8; v++) {
                int kk = ((v & 4) ? 16 : 0) + kb8 + 2 * (v & 3);
                a[mi].u32[v] = *(const unsigned int*)&sA[(mo + mi * 16 + r) * 40 + kk];
            }
#pragma unroll
        for (int ni = 0; ni < 2; ni++)
#pragma unroll
            for (int v = 0; v < 8; v++)
                b[ni].u32[v] = *(const unsigned int*)&sBt[(no + ni * 16 + r) * 36 + ((v & 4) ? 16 : 0) + kb8 + 2 * (v & 3)];

#pragma unroll
        for (int mi = 0; mi < 4; mi++)
#pragma unroll
            for (int ni = 0; ni < 2; ni++)
                acc[mi][ni] = WMMA_BF16(a[mi].v, b[ni].v, acc[mi][ni]);
        __syncthreads();
    }

    // epilogue: C layout -> lane holds col r, rows v + 8*hi per 16x16 tile
#pragma unroll
    for (int mi = 0; mi < 4; mi++)
#pragma unroll
        for (int ni = 0; ni < 2; ni++)
#pragma unroll
            for (int v = 0; v < 8; v++) {
                int gm = bm * 128 + mo + mi * 16 + v + 8 * hi;
                int gn = bn * 128 + no + ni * 16 + r;
                if (gm < M) {
                    float val = acc[mi][ni][v] + bias[gn];
                    if constexpr (EPI == EPI_F32_RES) {
                        outF[(size_t)gm * N + gn] = val + res[(size_t)gm * N + gn];
                    } else if constexpr (EPI == EPI_GELU_BF16) {
                        float ge = 0.5f * val * (1.0f + erff(val * 0.70710678118654752f));
                        outB[(size_t)gm * N + gn] = f2bf(ge);
                    } else {
                        outB[(size_t)gm * N + gn] = f2bf(val);
                    }
                }
            }
}

// ---------------------------------------------------------------------------
// Ragged flash attention. One block (4 waves) per (seq, head, 64-query tile).
// qkv: bf16 [total][3*768]; out: bf16 [total][768]
// ---------------------------------------------------------------------------
__global__ void __launch_bounds__(128) flash_attn_kernel(
    const u16* __restrict__ qkv, const int* __restrict__ cu,
    u16* __restrict__ outA) {
    const int D3 = 2304, DD = 768;
    int b = blockIdx.z, h = blockIdx.y;
    int s0 = cu[b];
    int len = cu[b + 1] - s0;
    int qt = blockIdx.x * 64;
    if (qt >= len) return;                       // block-uniform

    int tid = threadIdx.x, wave = tid >> 5, lane = tid & 31;
    int r = lane & 15, hi = lane >> 4;
    int kb8 = hi * 8, kb16 = hi * 16;

    __shared__ u16 ldsK[32 * 72];                // [key][d]   (16B-aligned rows)
    __shared__ u16 ldsVt[64 * 34];               // [d][key]
    __shared__ u16 ldsP[4][16 * 34];             // per-wave P [row][key]

    // ---- Q fragments (16 rows x 64 d = 2 A-frags), clamped like reference
    int qRow0 = qt + wave * 16;
    int qr = qRow0 + r; if (qr > len - 1) qr = len - 1;
    const u16* qp = qkv + (size_t)(s0 + qr) * D3 + h * 64;
    Frag aq[2];
#pragma unroll
    for (int s = 0; s < 2; s++)
#pragma unroll
        for (int v = 0; v < 8; v++) {
            int kk = s * 32 + ((v & 4) ? 16 : 0) + kb8 + 2 * (v & 3);
            aq[s].u32[v] = *(const unsigned int*)(qp + kk);
        }

    floatx8 o[4];
#pragma unroll
    for (int t = 0; t < 4; t++)
#pragma unroll
        for (int v = 0; v < 8; v++) o[t][v] = 0.0f;
    float mx[8], ls[8];
#pragma unroll
    for (int v = 0; v < 8; v++) { mx[v] = -3e38f; ls[v] = 0.0f; }

    const float scale = 0.125f;                  // 1/sqrt(64)
    int nk = (len + 31) >> 5;
    for (int kt = 0; kt < nk; ++kt) {
        // ---- stage K (row-major) and V (transposed) in LDS
#pragma unroll
        for (int it = 0; it < 2; ++it) {
            int c = tid + it * 128;              // 256 chunks of 8 bf16
            int kl = c >> 3, dc = (c & 7) * 8;
            int kg = kt * 32 + kl; if (kg > len - 1) kg = len - 1;
            const u16* kp = qkv + (size_t)(s0 + kg) * D3 + DD + h * 64 + dc;
            *(uint4*)&ldsK[kl * 72 + dc] = *(const uint4*)kp;
            uint4 tv = *(const uint4*)(kp + DD); // V
            unsigned int w[4] = {tv.x, tv.y, tv.z, tv.w};
#pragma unroll
            for (int i = 0; i < 4; i++) {
                ldsVt[(dc + 2 * i) * 34 + kl]     = (u16)(w[i] & 0xffffu);
                ldsVt[(dc + 2 * i + 1) * 34 + kl] = (u16)(w[i] >> 16);
            }
        }
        __syncthreads();

        // ---- S = Q * K^T : 2 key sub-tiles x 2 k-steps = 4 WMMAs
        floatx8 st[2];
#pragma unroll
        for (int t = 0; t < 2; t++) {
#pragma unroll
            for (int v = 0; v < 8; v++) st[t][v] = 0.0f;
#pragma unroll
            for (int s = 0; s < 2; s++) {
                Frag bk;
#pragma unroll
                for (int v = 0; v < 8; v++)
                    bk.u32[v] = *(const unsigned int*)&ldsK[(t * 16 + r) * 72 + s * 32 + ((v & 4) ? 16 : 0) + kb8 + 2 * (v & 3)];
                st[t] = WMMA_BF16(aq[s].v, bk.v, st[t]);
            }
        }

        // ---- online softmax (row = v + 8*hi, cols across 16-lane half)
        int key0 = kt * 32 + r;
        bool ok0 = key0 < len, ok1 = (key0 + 16) < len;
#pragma unroll
        for (int v = 0; v < 8; v++) {
            float s0v = ok0 ? st[0][v] * scale : -1e9f;
            float s1v = ok1 ? st[1][v] * scale : -1e9f;
            float m = fmaxf(s0v, s1v);
#pragma unroll
            for (int off = 1; off < 16; off <<= 1)
                m = fmaxf(m, __shfl_xor(m, off, 32));
            float mn = fmaxf(mx[v], m);
            float corr = __expf(mx[v] - mn);
            mx[v] = mn;
            float p0 = __expf(s0v - mn), p1 = __expf(s1v - mn);
            float rs = p0 + p1;
#pragma unroll
            for (int off = 1; off < 16; off <<= 1)
                rs += __shfl_xor(rs, off, 32);
            ls[v] = ls[v] * corr + rs;
#pragma unroll
            for (int t = 0; t < 4; t++) o[t][v] *= corr;
            int pr = v + 8 * hi;
            ldsP[wave][pr * 34 + r]      = f2bf(p0);
            ldsP[wave][pr * 34 + r + 16] = f2bf(p1);
        }

        // ---- O += P * V : 4 d-subtiles, K=32 keys -> 4 WMMAs
        Frag ap;
#pragma unroll
        for (int v = 0; v < 8; v++) {
            int kk = ((v & 4) ? 16 : 0) + kb8 + 2 * (v & 3);
            ap.u32[v] = *(const unsigned int*)&ldsP[wave][r * 34 + kk];
        }
#pragma unroll
        for (int t = 0; t < 4; t++) {
            Frag bv;
#pragma unroll
            for (int v = 0; v < 8; v++)
                bv.u32[v] = *(const unsigned int*)&ldsVt[(t * 16 + r) * 34 + ((v & 4) ? 16 : 0) + kb8 + 2 * (v & 3)];
            o[t] = WMMA_BF16(ap.v, bv.v, o[t]);
        }
        __syncthreads();
    }

    // ---- normalize + store bf16
#pragma unroll
    for (int v = 0; v < 8; v++) {
        float inv = 1.0f / ls[v];
        int row = qRow0 + v + 8 * hi;
        if (row < len) {
            u16* op = outA + (size_t)(s0 + row) * DD + h * 64;
#pragma unroll
            for (int t = 0; t < 4; t++) op[t * 16 + r] = f2bf(o[t][v] * inv);
        }
    }
}

// ---------------------------------------------------------------------------
// Host orchestration
// ---------------------------------------------------------------------------
extern "C" void kernel_launch(void* const* d_in, const int* in_sizes, int n_in,
                              void* d_out, int out_size, void* d_ws, size_t ws_size,
                              hipStream_t stream) {
    (void)n_in; (void)out_size; (void)ws_size;
    const float* x     = (const float*)d_in[0];
    const int*   cu    = (const int*)  d_in[1];
    const float* g1    = (const float*)d_in[2];
    const float* be1   = (const float*)d_in[3];
    const float* Wqkv  = (const float*)d_in[4];
    const float* bqkv  = (const float*)d_in[5];
    const float* Wo    = (const float*)d_in[6];
    const float* bo    = (const float*)d_in[7];
    const float* g2    = (const float*)d_in[8];
    const float* be2   = (const float*)d_in[9];
    const float* W1    = (const float*)d_in[10];
    const float* bfc1  = (const float*)d_in[11];
    const float* W2    = (const float*)d_in[12];
    const float* bfc2  = (const float*)d_in[13];

    const int D = 768, D3 = 2304, DFF = 3072, H = 12, B = 8, SMAX = 1024;
    const int total = in_sizes[0] / D;

    // workspace carve
    char* p = (char*)d_ws;
    auto carve = [&](size_t bytes) {
        void* q = (void*)p;
        p += (bytes + 255) & ~(size_t)255;
        return q;
    };
    u16*   wqkv_b = (u16*)carve((size_t)D * D3 * 2);
    u16*   wo_b   = (u16*)carve((size_t)D * D * 2);
    u16*   w1_b   = (u16*)carve((size_t)D * DFF * 2);
    u16*   w2_b   = (u16*)carve((size_t)DFF * D * 2);
    u16*   lnh    = (u16*)carve((size_t)total * D * 2);        // ln1 then ln2 output
    u16*   qkvm   = (u16*)carve((size_t)total * DFF * 2);      // qkv+attn, later m
    float* x2     = (float*)carve((size_t)total * D * 4);
    u16*   attn   = qkvm + (size_t)total * D3;

    auto cgrid = [](int n) { return dim3((n + 255) / 256); };
    cvt_bf16_kernel<<<cgrid(D * D3),  256, 0, stream>>>(Wqkv, wqkv_b, D * D3);
    cvt_bf16_kernel<<<cgrid(D * D),   256, 0, stream>>>(Wo,   wo_b,   D * D);
    cvt_bf16_kernel<<<cgrid(D * DFF), 256, 0, stream>>>(W1,   w1_b,   D * DFF);
    cvt_bf16_kernel<<<cgrid(DFF * D), 256, 0, stream>>>(W2,   w2_b,   DFF * D);

    int mBlks = (total + 127) / 128;

    // ln1
    ln_kernel<<<total, 256, 0, stream>>>(x, g1, be1, lnh);
    // qkv = ln1 @ Wqkv + bqkv   (bf16 out)
    gemm_bf16_kernel<EPI_BF16><<<dim3(D3 / 128, mBlks), 256, 0, stream>>>(
        lnh, wqkv_b, bqkv, nullptr, qkvm, nullptr, total, D3, D);
    // ragged flash attention
    flash_attn_kernel<<<dim3(SMAX / 64, H, B), 128, 0, stream>>>(qkvm, cu, attn);
    // x2 = x + attn @ Wo + bo   (fp32 out)
    gemm_bf16_kernel<EPI_F32_RES><<<dim3(D / 128, mBlks), 256, 0, stream>>>(
        attn, wo_b, bo, x, nullptr, x2, total, D, D);
    // ln2
    ln_kernel<<<total, 256, 0, stream>>>(x2, g2, be2, lnh);
    // m = gelu(ln2 @ W1 + bfc1) (bf16, overlays qkv/attn buffer)
    gemm_bf16_kernel<EPI_GELU_BF16><<<dim3(DFF / 128, mBlks), 256, 0, stream>>>(
        lnh, w1_b, bfc1, nullptr, qkvm, nullptr, total, DFF, D);
    // out = x2 + m @ W2 + bfc2  (fp32)
    gemm_bf16_kernel<EPI_F32_RES><<<dim3(D / 128, mBlks), 256, 0, stream>>>(
        qkvm, w2_b, bfc2, x2, nullptr, (float*)d_out, total, D, DFF);
}
